// DynamicTransformer_67808943670113
// MI455X (gfx1250) — matrix operand order repaired and checked by reading, not compiled, not run on hardware
//
#include <hip/hip_runtime.h>
#include <hip/hip_bf16.h>
#include <cstddef>

typedef __attribute__((ext_vector_type(16))) __bf16 v16bf;
typedef __attribute__((ext_vector_type(8)))  float  v8f;

// Problem constants (fixed by the reference)
#define BATCH 8
#define DIM   64
#define HID   128
#define IMGH  192
#define IMGW  192
#define HW    (IMGH * IMGW)      // 36864
#define TILE  16
#define HALO  18                 // TILE + 2
#define NPIX  (HALO * HALO)      // 324
#define NPAD  336                // next multiple of 16
#define XSTR  80                 // bf16 stride for x tile rows (160B, 32B aligned)
#define HSTR  144                // bf16 stride for h rows (288B, 32B aligned)
#define YSTR  144

// ---------------------------------------------------------------------------
// Kernel A: xbar[b,c] = mean over HW of x[b,c,:,:]   (linearity trick input)
// ---------------------------------------------------------------------------
__global__ void reduce_x(const float* __restrict__ x, float* __restrict__ xbar) {
    __shared__ float red[256];
    const int bc = blockIdx.x;                 // 0..511 == b*64+c
    const float* p = x + (size_t)bc * HW;
    float s = 0.f;
    for (int i = threadIdx.x; i < HW; i += 256) s += p[i];
    red[threadIdx.x] = s;
    __syncthreads();
    for (int off = 128; off > 0; off >>= 1) {
        if (threadIdx.x < off) red[threadIdx.x] += red[threadIdx.x + off];
        __syncthreads();
    }
    if (threadIdx.x == 0) xbar[bc] = red[0] * (1.0f / (float)HW);
}

// ---------------------------------------------------------------------------
// Kernel B: pack w_in / w_out into bf16 WMMA A-fragment lane order.
// 16-bit A 16x32 layout used here (contiguous k, not ISA table order):
//   lane<16 : row=m16+lane,    elems 0..15 -> K+0..15
//   lane>=16: row=m16+lane-16, elems 0..15 -> K+16..31
// ---------------------------------------------------------------------------
__device__ inline void pack_one(const float* __restrict__ W, int ldk,
                                __bf16* __restrict__ dst, int mt, int kt, int lane) {
    const int row = mt * 16 + (lane & 15);
    const int kb  = kt * 32 + ((lane & 16) ? 16 : 0);
    #pragma unroll
    for (int j = 0; j < 8; ++j) dst[j]     = (__bf16)W[row * ldk + kb + j];
    #pragma unroll
    for (int j = 0; j < 8; ++j) dst[8 + j] = (__bf16)W[row * ldk + kb + 8 + j];
}

__global__ void pack_weights(const float* __restrict__ w_in,
                             const float* __restrict__ w_out,
                             __bf16* __restrict__ winf,
                             __bf16* __restrict__ woutf) {
    // w_in: M=128 (8 mt), K=64 (2 kt) -> 512 fragments of 16 bf16
    // w_out: M=64 (4 mt), K=128 (4 kt) -> 512 fragments
    for (int f = threadIdx.x; f < 1024; f += 256) {
        if (f < 512) {
            const int lane = f & 31, fk = f >> 5;
            const int kt = fk & 1, mt = fk >> 1;
            pack_one(w_in, DIM, winf + f * 16, mt, kt, lane);
        } else {
            const int f2 = f - 512;
            const int lane = f2 & 31, fk = f2 >> 5;
            const int kt = fk & 3, mt = fk >> 2;
            pack_one(w_out, HID, woutf + f2 * 16, mt, kt, lane);
        }
    }
}

// ---------------------------------------------------------------------------
// Kernel C: per-sample dynamic 3x3 kernels from pooled stats.
//   g0 = w_in @ xbar + b_in   (valid because pooling commutes with 1x1 conv)
//   g1 = relu(wg1 @ g0 + bg1) ; ker = wg2 @ g1 + bg2
// ---------------------------------------------------------------------------
__global__ void gen_kernels(const float* __restrict__ xbar,
                            const float* __restrict__ w_in,
                            const float* __restrict__ b_in,
                            const float* __restrict__ wg1,
                            const float* __restrict__ bg1,
                            const float* __restrict__ wg2,
                            const float* __restrict__ bg2,
                            float* __restrict__ ker) {
    __shared__ float g0[HID];
    __shared__ float g1[HID];
    const int b = blockIdx.x;
    const int t = threadIdx.x;                 // 0..127
    const float* xb = xbar + b * DIM;

    float s = b_in[t];
    for (int c = 0; c < DIM; ++c) s += w_in[t * DIM + c] * xb[c];
    g0[t] = s;
    __syncthreads();

    s = bg1[t];
    for (int c = 0; c < HID; ++c) s += wg1[t * HID + c] * g0[c];
    g1[t] = s > 0.f ? s : 0.f;
    __syncthreads();

    for (int j = 0; j < 9; ++j) {
        const int i = t * 9 + j;
        float a = bg2[i];
        for (int c = 0; c < HID; ++c) a += wg2[i * HID + c] * g1[c];
        ker[b * HID * 9 + i] = a;
    }
}

// ---------------------------------------------------------------------------
// Kernel D: fused project_in (WMMA) -> dynamic depthwise 3x3 + LeakyReLU
//           -> project_out (WMMA). One 16x16 pixel tile per workgroup.
// ---------------------------------------------------------------------------
__global__ void __launch_bounds__(256)
fused_main(const float* __restrict__ x,
           const __bf16* __restrict__ winf,
           const __bf16* __restrict__ woutf,
           const float* __restrict__ b_in,
           const float* __restrict__ b_out,
           const float* __restrict__ ker,
           float* __restrict__ out) {
    extern __shared__ __align__(32) char smem[];
    __bf16* xs    = (__bf16*)smem;             // NPAD * XSTR  (x halo tile, bf16)
    __bf16* hs    = xs + NPAD * XSTR;          // NPAD * HSTR  (hidden, bf16)
    __bf16* ys    = hs + NPAD * HSTR;          // 256  * YSTR  (post-conv, bf16)
    float*  kersh = (float*)(ys + 256 * YSTR); // HID*9
    float*  binsh = kersh + HID * 9;           // HID
    float*  boutsh = binsh + HID;              // DIM

    const int tid  = threadIdx.x;
    const int lane = tid & 31;
    const int wave = tid >> 5;
    const int tx = blockIdx.x, ty = blockIdx.y, b = blockIdx.z;
    const int gx0 = tx * TILE - 1, gy0 = ty * TILE - 1;

    // warm L0/L2 for the small shared tables before the GEMMs need them
    __builtin_prefetch(winf + lane * 64, 0, 3);
    __builtin_prefetch(woutf + lane * 64, 0, 3);
    __builtin_prefetch(ker + b * HID * 9 + lane * 32, 0, 3);

    // stage 0: biases + per-sample 3x3 kernel into LDS
    if (tid < HID) binsh[tid]  = b_in[tid];
    if (tid < DIM) boutsh[tid] = b_out[tid];
    for (int i = tid; i < HID * 9; i += 256) kersh[i] = ker[b * HID * 9 + i];

    // stage 1: x halo tile -> LDS bf16, [pixel][channel], zero OOB / pad rows
    const float* xb = x + (size_t)b * DIM * HW;
    for (int idx = tid; idx < DIM * NPAD; idx += 256) {
        const int c = idx / NPAD;
        const int p = idx - c * NPAD;
        float v = 0.f;
        if (p < NPIX) {
            const int r = p / HALO, cc = p - r * HALO;
            const int gy = gy0 + r, gx = gx0 + cc;
            if (gy >= 0 && gy < IMGH && gx >= 0 && gx < IMGW)
                v = xb[(size_t)c * HW + gy * IMGW + gx];
        }
        xs[p * XSTR + c] = (__bf16)v;
    }
    __syncthreads();

    // stage 2: GEMM1  h = w_in @ x + b_in   M=128 N=336 K=64
    // mt == wave (8 waves cover M); 21 n-tiles per wave, 3 in flight so the
    // scheduler can fill WMMA->VALU hazard slots with independent chains.
    {
        const v16bf a0 = *(const v16bf*)(winf + ((size_t)((wave * 2 + 0) * 32 + lane)) * 16);
        const v16bf a1 = *(const v16bf*)(winf + ((size_t)((wave * 2 + 1) * 32 + lane)) * 16);
        for (int i0 = 0; i0 < 21; i0 += 3) {
            v8f acc[3];
            #pragma unroll
            for (int u = 0; u < 3; ++u) {
                const int nt = i0 + u;
                const __bf16* bp = xs + (nt * 16 + (lane & 15)) * XSTR + ((lane & 16) ? 16 : 0);
                v16bf b0 = *(const v16bf*)bp;
                v16bf b1 = *(const v16bf*)(bp + 32);
                v8f t = {0.f, 0.f, 0.f, 0.f, 0.f, 0.f, 0.f, 0.f};
                t = __builtin_amdgcn_wmma_f32_16x16x32_bf16(false, a0, false, b0, (short)0, t, false, false);
                t = __builtin_amdgcn_wmma_f32_16x16x32_bf16(false, a1, false, b1, (short)0, t, false, false);
                acc[u] = t;
            }
            #pragma unroll
            for (int u = 0; u < 3; ++u) {
                const int nt = i0 + u;
                const int N  = nt * 16 + (lane & 15);
                const int mb = wave * 16 + ((lane & 16) ? 8 : 0);
                #pragma unroll
                for (int v = 0; v < 8; ++v) {
                    const int m = mb + v;
                    hs[N * HSTR + m] = (__bf16)(acc[u][v] + binsh[m]);
                }
            }
        }
    }
    __syncthreads();

    // stage 3: per-sample depthwise 3x3 (SAME, zero pad) + LeakyReLU(0.1)
    // channel-pair vectorized: one 4B DS read = two bf16 taps.
    for (int idx = tid; idx < (HID / 2) * 256; idx += 256) {
        const int cp = idx & 63;               // channel pair 0..63
        const int q  = idx >> 6;               // pixel in 16x16 tile
        const int c  = cp * 2;
        const int oy = q >> 4, ox = q & 15;
        float a0 = 0.f, a1 = 0.f;
        #pragma unroll
        for (int dy = 0; dy < 3; ++dy) {
            const int gy = gy0 + oy + dy;
            if (gy < 0 || gy >= IMGH) continue;
            #pragma unroll
            for (int dx = 0; dx < 3; ++dx) {
                const int gx = gx0 + ox + dx;
                if (gx < 0 || gx >= IMGW) continue;
                const int p = (oy + dy) * HALO + (ox + dx);
                const unsigned hv = *(const unsigned*)(hs + p * HSTR + c);
                const float f0 = __uint_as_float(hv << 16);          // bf16 lo -> f32
                const float f1 = __uint_as_float(hv & 0xffff0000u);  // bf16 hi -> f32
                const int t = dy * 3 + dx;
                a0 += kersh[c * 9 + t] * f0;
                a1 += kersh[(c + 1) * 9 + t] * f1;
            }
        }
        const float y0 = a0 > 0.f ? a0 : 0.1f * a0;
        const float y1 = a1 > 0.f ? a1 : 0.1f * a1;
        union { __bf16 h[2]; unsigned u; } o;
        o.h[0] = (__bf16)y0;
        o.h[1] = (__bf16)y1;
        *(unsigned*)(ys + q * YSTR + c) = o.u;
    }
    __syncthreads();

    // stage 4: GEMM2  out = w_out @ y + b_out   M=64 N=256 K=128
    // mt == wave&3; 8 n-tiles per wave, 2 in flight.
    {
        const int mt = wave & 3;
        v16bf a[4];
        #pragma unroll
        for (int kt = 0; kt < 4; ++kt)
            a[kt] = *(const v16bf*)(woutf + ((size_t)((mt * 4 + kt) * 32 + lane)) * 16);

        float* ob = out + (size_t)b * DIM * HW;
        for (int i0 = 0; i0 < 8; i0 += 2) {
            v8f acc[2];
            #pragma unroll
            for (int u = 0; u < 2; ++u) {
                const int nt = 2 * (i0 + u) + (wave >> 2);
                const __bf16* bp = ys + (nt * 16 + (lane & 15)) * YSTR + ((lane & 16) ? 16 : 0);
                v8f t = {0.f, 0.f, 0.f, 0.f, 0.f, 0.f, 0.f, 0.f};
                #pragma unroll
                for (int kt = 0; kt < 4; ++kt) {
                    v16bf bm = *(const v16bf*)(bp + kt * 32);
                    t = __builtin_amdgcn_wmma_f32_16x16x32_bf16(false, a[kt], false, bm, (short)0, t, false, false);
                }
                acc[u] = t;
            }
            #pragma unroll
            for (int u = 0; u < 2; ++u) {
                const int nt = 2 * (i0 + u) + (wave >> 2);
                const int q  = nt * 16 + (lane & 15);
                const int oy = q >> 4, ox = q & 15;
                const int gy = ty * TILE + oy, gx = tx * TILE + ox;
                const int mb = mt * 16 + ((lane & 16) ? 8 : 0);
                #pragma unroll
                for (int v = 0; v < 8; ++v) {
                    const int m = mb + v;
                    ob[(size_t)m * HW + gy * IMGW + gx] = acc[u][v] + boutsh[m];
                }
            }
        }
    }
}

// ---------------------------------------------------------------------------
extern "C" void kernel_launch(void* const* d_in, const int* in_sizes, int n_in,
                              void* d_out, int out_size, void* d_ws, size_t ws_size,
                              hipStream_t stream) {
    (void)in_sizes; (void)n_in; (void)out_size; (void)ws_size;
    const float* x     = (const float*)d_in[0];
    const float* w_in  = (const float*)d_in[1];
    const float* b_in  = (const float*)d_in[2];
    const float* wg1   = (const float*)d_in[3];
    const float* bg1   = (const float*)d_in[4];
    const float* wg2   = (const float*)d_in[5];
    const float* bg2   = (const float*)d_in[6];
    const float* w_out = (const float*)d_in[7];
    const float* b_out = (const float*)d_in[8];
    float* out = (float*)d_out;

    // workspace carve (all 32B aligned)
    char* ws = (char*)d_ws;
    float*  xbar  = (float*)ws;                          // 512 f32
    float*  ker   = (float*)(ws + 2048);                 // 8*1152 f32
    __bf16* winf  = (__bf16*)(ws + 2048 + 36864);        // 8192 bf16
    __bf16* woutf = (__bf16*)(ws + 2048 + 36864 + 16384);// 8192 bf16

    reduce_x<<<dim3(BATCH * DIM), dim3(256), 0, stream>>>(x, xbar);
    pack_weights<<<dim3(1), dim3(256), 0, stream>>>(w_in, w_out, winf, woutf);
    gen_kernels<<<dim3(BATCH), dim3(HID), 0, stream>>>(xbar, w_in, b_in, wg1, bg1,
                                                       wg2, bg2, ker);

    const size_t smem =
        (size_t)(NPAD * XSTR + NPAD * HSTR + 256 * YSTR) * sizeof(__bf16) +
        (size_t)(HID * 9 + HID + DIM) * sizeof(float);   // ~224 KB (<320 KB WGP)
    (void)hipFuncSetAttribute((const void*)fused_main,
                              hipFuncAttributeMaxDynamicSharedMemorySize,
                              (int)smem);
    fused_main<<<dim3(IMGW / TILE, IMGH / TILE, BATCH), dim3(256), smem, stream>>>(
        x, winf, woutf, b_in, b_out, ker, out);
}
